// droDMTestNet_70471823393512
// MI455X (gfx1250) — compile-verified, matrix operand order checked by reading, not run on hardware
//
#include <hip/hip_runtime.h>

// ---------------------------------------------------------------------------
// CDNA5 (gfx1250) implementation of the Drosophila mushroom-body SNN step.
// All GEMMs run on v_wmma_f32_16x16x32_f16 (FP16 A/B, FP32 accumulate).
// Spikes are {0,1} -> exact in f16. Membranes & STDP traces stay in f32.
// ---------------------------------------------------------------------------

typedef _Float16 f16;
typedef __attribute__((ext_vector_type(16))) _Float16 v16h;
typedef __attribute__((ext_vector_type(8)))  float    v8f;

#define THRESH 0.5f
#define DECAY  0.99f

// Network dims
#define TSTEPS 32
#define BATCH  128
#define N_IN   512
#define N_VIS  512
#define N_KC   4096
#define N_MBON 128
#define N_APL  64
#define N_DA   64
#define N_DAIN 64

__device__ __forceinline__ v8f wmma16(v16h a, v16h b, v8f c) {
  // D = A(16x32 f16) * B(32x16 f16) + C(16x16 f32)
  return __builtin_amdgcn_wmma_f32_16x16x32_f16(
      /*neg_a=*/false, a, /*neg_b=*/false, b,
      /*c_mod=*/(short)0, c, /*reuse_a=*/false, /*reuse_b=*/false);
}

// A fragment loader, row-major f16 matrix M[rows, ld], tile (m0,k0).
// ISA A-layout (16-bit, 16x32): lanes 0-15 row=m0+lane, K={0..7,16..23};
// lanes 16-31 row=m0+lane-16, K={8..15,24..31}.
// The SAME loader serves the B operand when applied to B^T (weights are
// stored pre-transposed [N,K] so both operand loads are contiguous b128s).
__device__ __forceinline__ v16h ldA(const f16* __restrict__ M, int ld,
                                    int m0, int k0, int lane) {
  const int h = (lane >> 4) & 1;
  const int r = lane & 15;
  const f16* p = M + (size_t)(m0 + r) * ld + (k0 + h * 8);
  v16h v;
#pragma unroll
  for (int i = 0; i < 8; ++i) v[i] = p[i];
#pragma unroll
  for (int i = 0; i < 8; ++i) v[8 + i] = p[16 + i];
  return v;
}

// B fragment from a row-major f16 matrix S[K, ld] (spike matrices, not
// pre-transposed): lane = column n0+(lane&15), K split as above (strided).
__device__ __forceinline__ v16h ldB_rm(const f16* __restrict__ S, int ld,
                                       int k0, int n0, int lane) {
  const int h = (lane >> 4) & 1;
  const int c = lane & 15;
  const f16* p = S + (size_t)(k0 + h * 8) * ld + n0 + c;
  v16h v;
#pragma unroll
  for (int i = 0; i < 8; ++i) v[i] = p[(size_t)i * ld];
#pragma unroll
  for (int i = 0; i < 8; ++i) v[8 + i] = p[(size_t)(16 + i) * ld];
  return v;
}

// A fragment of Tr^T where Tr is row-major f32 [K, ldm] (the STDP traces):
// A[m,k] = Tr[k,m], converted f32->f16 on the fly.
__device__ __forceinline__ v16h ldA_T_f32(const float* __restrict__ Tr, int ldm,
                                          int m0, int k0, int lane) {
  const int h = (lane >> 4) & 1;
  const int r = lane & 15;
  const float* p = Tr + (size_t)(k0 + h * 8) * ldm + (m0 + r);
  v16h v;
#pragma unroll
  for (int i = 0; i < 8; ++i) v[i] = (f16)p[(size_t)i * ldm];
#pragma unroll
  for (int i = 0; i < 8; ++i) v[8 + i] = (f16)p[(size_t)(16 + i) * ldm];
  return v;
}

// IF epilogue: C/D layout => acc[j] is (row m0+j+8*(lane>=16), col n0+(lane&15))
__device__ __forceinline__ void if_store(v8f acc, float* __restrict__ V,
                                         f16* __restrict__ S, int ld,
                                         int m0, int n0, int lane) {
  const int h = (lane >> 4) & 1;
  const int c = lane & 15;
#pragma unroll
  for (int j = 0; j < 8; ++j) {
    const size_t idx = (size_t)(m0 + j + 8 * h) * ld + n0 + c;
    float v = V[idx] + acc[j];
    float s = (v >= THRESH) ? 1.0f : 0.0f;
    V[idx] = v * (1.0f - s);
    S[idx] = (f16)s;
  }
}

// ---------------------------------------------------------------------------
// Stage kernels (block = 32x4 : 4 wave32 per block, one 16x16 tile per wave)
// ---------------------------------------------------------------------------

__global__ __launch_bounds__(128) void k_vis(const f16* __restrict__ X,
                                             const f16* __restrict__ W0t,
                                             float* __restrict__ v0,
                                             f16* __restrict__ s_vis) {
  const int lane = threadIdx.x;
  const int n0 = (blockIdx.x * 4 + threadIdx.y) * 16;
  const int m0 = blockIdx.y * 16;
  v8f acc = {};
  for (int k0 = 0; k0 < N_IN; k0 += 32)
    acc = wmma16(ldA(X, N_IN, m0, k0, lane), ldA(W0t, N_IN, n0, k0, lane), acc);
  if_store(acc, v0, s_vis, N_VIS, m0, n0, lane);
}

__global__ __launch_bounds__(128) void k_kc(const f16* __restrict__ s_vis,
                                            const f16* __restrict__ W1t,
                                            const f16* __restrict__ oA,
                                            const f16* __restrict__ W5t,
                                            float* __restrict__ v1,
                                            f16* __restrict__ s_kc) {
  const int lane = threadIdx.x;
  const int n0 = (blockIdx.x * 4 + threadIdx.y) * 16;
  const int m0 = blockIdx.y * 16;
  v8f acc = {};
  for (int k0 = 0; k0 < N_VIS; k0 += 32)
    acc = wmma16(ldA(s_vis, N_VIS, m0, k0, lane), ldA(W1t, N_VIS, n0, k0, lane), acc);
  for (int k0 = 0; k0 < N_APL; k0 += 32)
    acc = wmma16(ldA(oA, N_APL, m0, k0, lane), ldA(W5t, N_APL, n0, k0, lane), acc);
  if_store(acc, v1, s_kc, N_KC, m0, n0, lane);
}

__global__ __launch_bounds__(128) void k_mbon(const f16* __restrict__ s_kc,
                                              const f16* __restrict__ W2t,
                                              const f16* __restrict__ oM,
                                              const f16* __restrict__ W3t,
                                              const f16* __restrict__ oD,
                                              const f16* __restrict__ W9t,
                                              float* __restrict__ v2,
                                              f16* __restrict__ s_m) {
  const int lane = threadIdx.x;
  const int n0 = (blockIdx.x * 4 + threadIdx.y) * 16;
  const int m0 = blockIdx.y * 16;
  v8f acc = {};
  for (int k0 = 0; k0 < N_KC; k0 += 32)
    acc = wmma16(ldA(s_kc, N_KC, m0, k0, lane), ldA(W2t, N_KC, n0, k0, lane), acc);
  for (int k0 = 0; k0 < N_MBON; k0 += 32)
    acc = wmma16(ldA(oM, N_MBON, m0, k0, lane), ldA(W3t, N_MBON, n0, k0, lane), acc);
  for (int k0 = 0; k0 < N_DA; k0 += 32)
    acc = wmma16(ldA(oD, N_DA, m0, k0, lane), ldA(W9t, N_DA, n0, k0, lane), acc);
  if_store(acc, v2, s_m, N_MBON, m0, n0, lane);
}

__global__ __launch_bounds__(128) void k_apl(const f16* __restrict__ s_kc,
                                             const f16* __restrict__ W4t,
                                             const f16* __restrict__ oD,
                                             const f16* __restrict__ W6t,
                                             float* __restrict__ v3,
                                             f16* __restrict__ s_a) {
  const int lane = threadIdx.x;
  const int n0 = (blockIdx.x * 4 + threadIdx.y) * 16;
  const int m0 = blockIdx.y * 16;
  v8f acc = {};
  for (int k0 = 0; k0 < N_KC; k0 += 32)
    acc = wmma16(ldA(s_kc, N_KC, m0, k0, lane), ldA(W4t, N_KC, n0, k0, lane), acc);
  for (int k0 = 0; k0 < N_DA; k0 += 32)
    acc = wmma16(ldA(oD, N_DA, m0, k0, lane), ldA(W6t, N_DA, n0, k0, lane), acc);
  if_store(acc, v3, s_a, N_APL, m0, n0, lane);
}

__global__ __launch_bounds__(128) void k_da(const f16* __restrict__ s_a,
                                            const f16* __restrict__ W7t,
                                            const f16* __restrict__ Xda,
                                            const f16* __restrict__ W8t,
                                            float* __restrict__ v4,
                                            f16* __restrict__ s_d) {
  const int lane = threadIdx.x;
  const int n0 = (blockIdx.x * 4 + threadIdx.y) * 16;
  const int m0 = blockIdx.y * 16;
  v8f acc = {};
  for (int k0 = 0; k0 < N_APL; k0 += 32)
    acc = wmma16(ldA(s_a, N_APL, m0, k0, lane), ldA(W7t, N_APL, n0, k0, lane), acc);
  for (int k0 = 0; k0 < N_DAIN; k0 += 32)
    acc = wmma16(ldA(Xda, N_DAIN, m0, k0, lane), ldA(W8t, N_DAIN, n0, k0, lane), acc);
  if_store(acc, v4, s_d, N_DA, m0, n0, lane);
}

// dw = Tr^T @ S  (Tr: f32 [128, Mdim], S: f16 [128, Ndim], Out: f32 [Mdim, Ndim])
__global__ __launch_bounds__(128) void k_dw(const float* __restrict__ Tr, int Mdim,
                                            const f16* __restrict__ S, int Ndim,
                                            float* __restrict__ Out) {
  const int lane = threadIdx.x;
  const int n0 = (blockIdx.x * 4 + threadIdx.y) * 16;
  const int m0 = blockIdx.y * 16;
  v8f acc = {};
  for (int k0 = 0; k0 < BATCH; k0 += 32)
    acc = wmma16(ldA_T_f32(Tr, Mdim, m0, k0, lane), ldB_rm(S, Ndim, k0, n0, lane), acc);
  const int h = (lane >> 4) & 1;
  const int c = lane & 15;
#pragma unroll
  for (int j = 0; j < 8; ++j)
    Out[(size_t)(m0 + j + 8 * h) * Ndim + n0 + c] = acc[j];
}

// ---------------------------------------------------------------------------
// Small elementwise kernels
// ---------------------------------------------------------------------------

__global__ void k_trace(float* __restrict__ tr, const f16* __restrict__ s, int n) {
  int i = blockIdx.x * blockDim.x + threadIdx.x;
  if (i < n) tr[i] = tr[i] * DECAY + (float)s[i];
}

__global__ void k_cvt(const float* __restrict__ src, f16* __restrict__ dst, int n) {
  int i = blockIdx.x * blockDim.x + threadIdx.x;
  if (i < n) dst[i] = (f16)src[i];
}

// src f32 [K,N] -> dst f16 [N,K] (transposed weight panel for contiguous B loads)
__global__ void k_cvtT(const float* __restrict__ src, f16* __restrict__ dst,
                       int K, int N) {
  int i = blockIdx.x * blockDim.x + threadIdx.x;
  if (i >= K * N) return;
  int k = i / N, n = i % N;
  dst[(size_t)n * K + k] = (f16)src[i];
}

__global__ void k_zero16(uint4* __restrict__ p, int n16) {
  int i = blockIdx.x * blockDim.x + threadIdx.x;
  if (i < n16) p[i] = make_uint4(0u, 0u, 0u, 0u);
}

__global__ void k_f16f32(const f16* __restrict__ s, float* __restrict__ d, int n) {
  int i = blockIdx.x * blockDim.x + threadIdx.x;
  if (i < n) d[i] = (float)s[i];
}

// ---------------------------------------------------------------------------
// Host orchestration
// ---------------------------------------------------------------------------

extern "C" void kernel_launch(void* const* d_in, const int* in_sizes, int n_in,
                              void* d_out, int out_size, void* d_ws, size_t ws_size,
                              hipStream_t stream) {
  (void)in_sizes; (void)n_in; (void)out_size; (void)ws_size;

  const float* inp    = (const float*)d_in[0];   // [T,B,N_IN]
  const float* inp_da = (const float*)d_in[1];   // [T,B,N_DAIN]
  const float* W[10];
  for (int i = 0; i < 10; ++i) W[i] = (const float*)d_in[2 + i];

  // --- workspace bump allocator (256B aligned) ---
  size_t off = 0;
  char* base = (char*)d_ws;
  auto alloc = [&](size_t bytes) -> char* {
    char* p = base + off;
    off += (bytes + 255) & ~(size_t)255;
    return p;
  };

  // Pre-converted constants (f16); weights stored transposed [out,in].
  f16* W0t = (f16*)alloc((size_t)N_VIS * N_IN   * 2);
  f16* W1t = (f16*)alloc((size_t)N_KC  * N_VIS  * 2);
  f16* W2t = (f16*)alloc((size_t)N_MBON* N_KC   * 2);
  f16* W3t = (f16*)alloc((size_t)N_MBON* N_MBON * 2);
  f16* W4t = (f16*)alloc((size_t)N_APL * N_KC   * 2);
  f16* W5t = (f16*)alloc((size_t)N_KC  * N_APL  * 2);
  f16* W6t = (f16*)alloc((size_t)N_APL * N_DA   * 2);
  f16* W7t = (f16*)alloc((size_t)N_DA  * N_APL  * 2);
  f16* W8t = (f16*)alloc((size_t)N_DA  * N_DAIN * 2);
  f16* W9t = (f16*)alloc((size_t)N_MBON* N_DA   * 2);
  f16* Xh  = (f16*)alloc((size_t)TSTEPS * BATCH * N_IN   * 2);
  f16* XDh = (f16*)alloc((size_t)TSTEPS * BATCH * N_DAIN * 2);

  // Per-step live buffers (fully rewritten before read each step).
  f16* s_vis = (f16*)alloc((size_t)BATCH * N_VIS * 2);
  f16* s_kc  = (f16*)alloc((size_t)BATCH * N_KC  * 2);

  // Persistent state (zeroed below).
  char* state0 = base + off;
  float* v0     = (float*)alloc((size_t)BATCH * N_VIS  * 4);
  float* v1     = (float*)alloc((size_t)BATCH * N_KC   * 4);
  float* v2     = (float*)alloc((size_t)BATCH * N_MBON * 4);
  float* v3     = (float*)alloc((size_t)BATCH * N_APL  * 4);
  float* v4     = (float*)alloc((size_t)BATCH * N_DA   * 4);
  float* tr_apl = (float*)alloc((size_t)BATCH * N_APL  * 4);  // t_apl_kc
  float* tr_kc  = (float*)alloc((size_t)BATCH * N_KC   * 4);  // t_kc_apl
  f16* sm[2] = { (f16*)alloc((size_t)BATCH * N_MBON * 2), (f16*)alloc((size_t)BATCH * N_MBON * 2) };
  f16* sa[2] = { (f16*)alloc((size_t)BATCH * N_APL  * 2), (f16*)alloc((size_t)BATCH * N_APL  * 2) };
  f16* sd[2] = { (f16*)alloc((size_t)BATCH * N_DA   * 2), (f16*)alloc((size_t)BATCH * N_DA   * 2) };
  size_t state_bytes = (size_t)((base + off) - state0);

  // --- one-time prep (inside launch: deterministic, graph-capture safe) ---
  {
    int n, g;
    n = TSTEPS * BATCH * N_IN;  g = (n + 255) / 256;
    k_cvt<<<g, 256, 0, stream>>>(inp, Xh, n);
    n = TSTEPS * BATCH * N_DAIN; g = (n + 255) / 256;
    k_cvt<<<g, 256, 0, stream>>>(inp_da, XDh, n);

    auto cvtT = [&](const float* s, f16* d, int K, int N) {
      int m = K * N;
      k_cvtT<<<(m + 255) / 256, 256, 0, stream>>>(s, d, K, N);
    };
    cvtT(W[0], W0t, N_IN,   N_VIS);
    cvtT(W[1], W1t, N_VIS,  N_KC);
    cvtT(W[2], W2t, N_KC,   N_MBON);
    cvtT(W[3], W3t, N_MBON, N_MBON);
    cvtT(W[4], W4t, N_KC,   N_APL);
    cvtT(W[5], W5t, N_APL,  N_KC);
    cvtT(W[6], W6t, N_DA,   N_APL);
    cvtT(W[7], W7t, N_APL,  N_DA);
    cvtT(W[8], W8t, N_DAIN, N_DA);
    cvtT(W[9], W9t, N_DA,   N_MBON);

    int n16 = (int)(state_bytes / 16);
    k_zero16<<<(n16 + 255) / 256, 256, 0, stream>>>((uint4*)state0, n16);
  }

  // --- timestep loop ---
  const dim3 blk(32, 4);
  for (int t = 0; t < TSTEPS; ++t) {
    const int p = t & 1, q = 1 - p;   // prev / current spike buffers
    const f16* Xt  = Xh  + (size_t)t * BATCH * N_IN;
    const f16* Xdt = XDh + (size_t)t * BATCH * N_DAIN;

    // VIS: s_vis = IF(v0, x@W0)
    k_vis<<<dim3(N_VIS / 64, BATCH / 16), blk, 0, stream>>>(Xt, W0t, v0, s_vis);
    // KC: s_kc = IF(v1, s_vis@W1 + oA@W5)
    k_kc<<<dim3(N_KC / 64, BATCH / 16), blk, 0, stream>>>(s_vis, W1t, sa[p], W5t, v1, s_kc);
    // t_apl_kc = t_apl_kc*DECAY + oA
    k_trace<<<(BATCH * N_APL + 255) / 256, 256, 0, stream>>>(tr_apl, sa[p], BATCH * N_APL);
    // MBON: IF(v2, s_kc@W2 + oM@W3 + oD@W9)
    k_mbon<<<dim3(N_MBON / 64, BATCH / 16), blk, 0, stream>>>(s_kc, W2t, sm[p], W3t, sd[p], W9t, v2, sm[q]);
    // APL: IF(v3, s_kc@W4 + oD@W6)
    k_apl<<<dim3(N_APL / 64, BATCH / 16), blk, 0, stream>>>(s_kc, W4t, sd[p], W6t, v3, sa[q]);
    // t_kc_apl = t_kc_apl*DECAY + s_kc
    k_trace<<<(BATCH * N_KC + 255) / 256, 256, 0, stream>>>(tr_kc, s_kc, BATCH * N_KC);
    // DA: IF(v4, s_a(new)@W7 + xda@W8)
    k_da<<<dim3(N_DA / 64, BATCH / 16), blk, 0, stream>>>(sa[q], W7t, Xdt, W8t, v4, sd[q]);
  }

  // After t=31: p=1, current buffers are index 0.
  float* out_sm  = (float*)d_out;                          // [128,128]
  float* out_dw5 = out_sm  + BATCH * N_MBON;               // [64,4096]
  float* out_dw4 = out_dw5 + (size_t)N_APL * N_KC;         // [4096,64]

  k_f16f32<<<(BATCH * N_MBON + 255) / 256, 256, 0, stream>>>(sm[0], out_sm, BATCH * N_MBON);
  // dw5 = t_apl_kc^T @ s_kc : [64,4096]
  k_dw<<<dim3(N_KC / 64, N_APL / 16), blk, 0, stream>>>(tr_apl, N_APL, s_kc, N_KC, out_dw5);
  // dw4 = t_kc_apl^T @ s_a : [4096,64]
  k_dw<<<dim3(N_APL / 64, N_KC / 16), blk, 0, stream>>>(tr_kc, N_KC, sa[0], N_APL, out_dw4);
}
